// SACLRAll_68109591380641
// MI455X (gfx1250) — matrix operand order repaired and checked by reading, not compiled, not run on hardware
//
#include <hip/hip_runtime.h>
#include <hip/hip_bf16.h>

typedef __attribute__((ext_vector_type(16))) _Float16 v16h;
typedef __attribute__((ext_vector_type(8)))  _Float16 v8h;
typedef __attribute__((ext_vector_type(4)))  _Float16 v4h;
typedef __attribute__((ext_vector_type(8)))  float    v8f;
typedef __attribute__((ext_vector_type(4)))  float    v4f;

#define BHALF   4096      // B
#define TWOB    8192      // 2B
#define DDIM    128       // D
#define NDATA   1281167.0f
#define CPN     128       // column patches (64 cols each) => partial row stride
#define NTRI    8256      // 128*129/2 upper-triangular patch tasks

// ---------------------------------------------------------------------------
// Kernel 1: L2-normalize rows of feats (fp32), store unit rows as f16.
// One wave (32 lanes) per row; each lane handles 4 contiguous floats.
// ---------------------------------------------------------------------------
__global__ __launch_bounds__(256) void saclr_normalize(const float* __restrict__ feats,
                                                       _Float16* __restrict__ Fn) {
    const int lane = threadIdx.x & 31;
    const int row  = blockIdx.x * 8 + (threadIdx.x >> 5);   // 1024 blocks * 8 waves = 8192 rows

    const v4f v = *((const v4f*)(feats + (size_t)row * DDIM) + lane);
    float ss = v[0]*v[0] + v[1]*v[1] + v[2]*v[2] + v[3]*v[3];
    ss += __shfl_xor(ss, 16);
    ss += __shfl_xor(ss, 8);
    ss += __shfl_xor(ss, 4);
    ss += __shfl_xor(ss, 2);
    ss += __shfl_xor(ss, 1);
    const float rn = rsqrtf(ss);

    v4h h;
    h[0] = (_Float16)(v[0] * rn);
    h[1] = (_Float16)(v[1] * rn);
    h[2] = (_Float16)(v[2] * rn);
    h[3] = (_Float16)(v[3] * rn);
    *((v4h*)(Fn + (size_t)row * DDIM) + lane) = h;
}

// ---------------------------------------------------------------------------
// Kernel 2: exact fp32 normalized dot diag[i] = <fa_i, fb_i>/(|fa_i||fb_i|).
// One wave per i.
// ---------------------------------------------------------------------------
__global__ __launch_bounds__(256) void saclr_diag(const float* __restrict__ feats,
                                                  float* __restrict__ diag) {
    const int lane = threadIdx.x & 31;
    const int i    = blockIdx.x * 8 + (threadIdx.x >> 5);   // 512 blocks -> 4096

    const v4f a = *((const v4f*)(feats + (size_t)i * DDIM) + lane);
    const v4f b = *((const v4f*)(feats + (size_t)(i + BHALF) * DDIM) + lane);
    float sa = a[0]*a[0] + a[1]*a[1] + a[2]*a[2] + a[3]*a[3];
    float sb = b[0]*b[0] + b[1]*b[1] + b[2]*b[2] + b[3]*b[3];
    float sc = a[0]*b[0] + a[1]*b[1] + a[2]*b[2] + a[3]*b[3];
#pragma unroll
    for (int off = 16; off >= 1; off >>= 1) {
        sa += __shfl_xor(sa, off);
        sb += __shfl_xor(sb, off);
        sc += __shfl_xor(sc, off);
    }
    if (lane == 0) diag[i] = sc * rsqrtf(sa * sb);
}

// ---------------------------------------------------------------------------
// Kernel 3: symmetric WMMA Gram tiles + exp + masked row/col sums.
// Tasks cover only the upper triangle (rp <= cp): 8256 wave-tasks instead of
// 16384 -> halves WMMA and exp work. Off-diagonal patches also emit their
// column sums (= transpose row sums) so every partial[row][colPatch] entry
// is written exactly once (deterministic, no atomics).
// ---------------------------------------------------------------------------
__device__ __forceinline__ v16h load_a_frag(const _Float16* __restrict__ Fn,
                                            int row, int k0, int hi) {
    // 16-bit A-matrix 16x32 layout: elems 0..7 = K{k0 + hi*8 + 0..7},
    //                               elems 8..15 = K{k0 + 16 + hi*8 + 0..7}
    const _Float16* p = Fn + (size_t)row * DDIM + k0 + hi * 8;
    const v8h lo = *(const v8h*)(p);
    const v8h hi8 = *(const v8h*)(p + 16);
    v16h r;
#pragma unroll
    for (int i = 0; i < 8; ++i) { r[i] = lo[i]; r[i + 8] = hi8[i]; }
    return r;
}

__device__ __forceinline__ v16h load_b_frag(const _Float16* __restrict__ Fn,
                                            int col, int k0, int hi) {
    // 16-bit B-matrix 32x16 layout: elems 0..7 = K{k0 + hi*8 + 0..7},
    //                               elems 8..15 = K{k0 + 16 + hi*8 + 0..7}
    const _Float16* p = Fn + (size_t)col * DDIM + k0 + hi * 8;
    const v8h lo = *(const v8h*)(p);
    const v8h hi8 = *(const v8h*)(p + 16);
    v16h r;
#pragma unroll
    for (int i = 0; i < 8; ++i) { r[i] = lo[i]; r[i + 8] = hi8[i]; }
    return r;
}

__global__ __launch_bounds__(256) void saclr_gemm_exp(const _Float16* __restrict__ Fn,
                                                      float* __restrict__ partial) {
    const int lane = threadIdx.x & 31;
    const int wave = threadIdx.x >> 5;
    const int task = blockIdx.x * 8 + wave;        // 1032 blocks * 8 = 8256 tasks
    // Decode linear task -> (rp, cp) with rp <= cp over a 128x128 patch grid.
    // base(i) = i*128 - i*(i-1)/2 = #tasks with first index < i.
    int rp = (int)((257.0f - sqrtf(66049.0f - 8.0f * (float)task)) * 0.5f);
    if (rp > 127) rp = 127;
    while (rp > 0 && (rp * 128 - (rp * (rp - 1)) / 2) > task) --rp;
    while (((rp + 1) * 128 - ((rp + 1) * rp) / 2) <= task) ++rp;
    const int cp = rp + (task - (rp * 128 - (rp * (rp - 1)) / 2));

    const int rbase = rp * 64;
    const int cbase = cp * 64;
    const int ln = lane & 15;
    const int hi = lane >> 4;
    const bool onDiag   = (rp == cp);
    const bool needMask = onDiag || (cp == (rp ^ 64));   // row^4096 shifts patch by ^64

    v8f acc[4][4];
#pragma unroll
    for (int i = 0; i < 4; ++i)
#pragma unroll
        for (int j = 0; j < 4; ++j) acc[i][j] = (v8f)(0.0f);

#pragma unroll
    for (int k0 = 0; k0 < DDIM; k0 += 32) {
        v16h a[4], b[4];
#pragma unroll
        for (int t = 0; t < 4; ++t) a[t] = load_a_frag(Fn, rbase + t * 16 + ln, k0, hi);
#pragma unroll
        for (int t = 0; t < 4; ++t) b[t] = load_b_frag(Fn, cbase + t * 16 + ln, k0, hi);
#pragma unroll
        for (int i = 0; i < 4; ++i)
#pragma unroll
            for (int j = 0; j < 4; ++j)
                acc[i][j] = __builtin_amdgcn_wmma_f32_16x16x32_f16(
                    false, a[i], false, b[j], (short)0, acc[i][j], false, false);
    }

    // e = exp(4s - 4), masked on the diagonal / cross-diagonal; accumulate
    // per-lane row sums (over this patch's columns) and column sums.
    float rowAcc[4][8];   // [ti][v] : lane-partial (this lane's column ln contribution)
    float colAcc[4];      // [tj]    : per-lane column (cbase + tj*16 + ln) partial
#pragma unroll
    for (int ti = 0; ti < 4; ++ti)
#pragma unroll
        for (int v = 0; v < 8; ++v) rowAcc[ti][v] = 0.0f;
#pragma unroll
    for (int tj = 0; tj < 4; ++tj) colAcc[tj] = 0.0f;

    if (needMask) {
#pragma unroll
        for (int ti = 0; ti < 4; ++ti)
#pragma unroll
            for (int v = 0; v < 8; ++v) {
                const int row = rbase + ti * 16 + v + hi * 8;
#pragma unroll
                for (int tj = 0; tj < 4; ++tj) {
                    const int col = cbase + tj * 16 + ln;
                    const bool masked = (col == row) || (col == (row ^ BHALF));
                    const float e = masked ? 0.0f : __expf(4.0f * acc[ti][tj][v] - 4.0f);
                    rowAcc[ti][v] += e;
                    colAcc[tj]    += e;
                }
            }
    } else {
#pragma unroll
        for (int ti = 0; ti < 4; ++ti)
#pragma unroll
            for (int v = 0; v < 8; ++v)
#pragma unroll
                for (int tj = 0; tj < 4; ++tj) {
                    const float e = __expf(4.0f * acc[ti][tj][v] - 4.0f);
                    rowAcc[ti][v] += e;
                    colAcc[tj]    += e;
                }
    }

    // Row sums: reduce each rowAcc across the 16-lane group.
#pragma unroll
    for (int ti = 0; ti < 4; ++ti) {
        float rs[8];
#pragma unroll
        for (int v = 0; v < 8; ++v) {
            float s = rowAcc[ti][v];
            s += __shfl_xor(s, 1);
            s += __shfl_xor(s, 2);
            s += __shfl_xor(s, 4);
            s += __shfl_xor(s, 8);
            rs[v] = s;
        }
        if (ln == 0) {
#pragma unroll
            for (int v = 0; v < 8; ++v) {
                const int row = rbase + ti * 16 + v + hi * 8;
                partial[(size_t)row * CPN + cp] = rs[v];
            }
        }
    }

    // Column sums (transpose contribution) for off-diagonal patches:
    // combine the two lane halves, then lanes of half 0 write one column each.
    if (!onDiag) {
#pragma unroll
        for (int tj = 0; tj < 4; ++tj) {
            float c = colAcc[tj];
            c += __shfl_xor(c, 16);
            if (hi == 0) {
                const int col = cbase + tj * 16 + ln;
                partial[(size_t)col * CPN + rp] = c;
            }
        }
    }
}

// ---------------------------------------------------------------------------
// Kernel 4: q[r] = sum over 128 column-patch partials (fixed order per lane).
// One wave per row.
// ---------------------------------------------------------------------------
__global__ __launch_bounds__(256) void saclr_rowsum(const float* __restrict__ partial,
                                                    float* __restrict__ q) {
    const int lane = threadIdx.x & 31;
    const int r    = blockIdx.x * 8 + (threadIdx.x >> 5);   // 1024 blocks -> 8192 rows
    const float* p = partial + (size_t)r * CPN;
    float s = p[lane] + p[lane + 32] + p[lane + 64] + p[lane + 96];
    s += __shfl_xor(s, 16);
    s += __shfl_xor(s, 8);
    s += __shfl_xor(s, 4);
    s += __shfl_xor(s, 2);
    s += __shfl_xor(s, 1);
    if (lane == 0) q[r] = s;
}

// ---------------------------------------------------------------------------
// Kernel 5: final deterministic reduction to the scalar loss.
// loss = mean_i(4 - 4*d_i) + mean_i( (q[i] + q[i+B]) / (2 * z_hat_i) )
// ---------------------------------------------------------------------------
__global__ __launch_bounds__(256) void saclr_final(const float* __restrict__ q,
                                                   const float* __restrict__ diag,
                                                   const float* __restrict__ s_inv,
                                                   const int* __restrict__ idx,
                                                   float* __restrict__ out) {
    __shared__ float sm[256];
    const int tid = threadIdx.x;
    const float zscale = (2.0f * (float)BHALF - 2.0f) / (NDATA * NDATA);
    float acc = 0.0f;
    for (int i = tid; i < BHALF; i += 256) {
        const float zh  = s_inv[idx[i]] * zscale;
        const float rep = (q[i] + q[i + BHALF]) / (2.0f * zh);
        acc += (4.0f - 4.0f * diag[i]) + rep;
    }
    sm[tid] = acc;
    __syncthreads();
#pragma unroll
    for (int s = 128; s > 0; s >>= 1) {
        if (tid < s) sm[tid] += sm[tid + s];
        __syncthreads();
    }
    if (tid == 0) out[0] = sm[0] / (float)BHALF;
}

// ---------------------------------------------------------------------------
// Host launcher
// ---------------------------------------------------------------------------
extern "C" void kernel_launch(void* const* d_in, const int* in_sizes, int n_in,
                              void* d_out, int out_size, void* d_ws, size_t ws_size,
                              hipStream_t stream) {
    const float* feats = (const float*)d_in[0];   // (2B, D) fp32
    const float* s_inv = (const float*)d_in[1];   // (N_DATA,) fp32
    const int*   fidx  = (const int*)d_in[2];     // (B,) int32
    float* out = (float*)d_out;

    char* ws = (char*)d_ws;
    _Float16* Fn      = (_Float16*)(ws);                                  // 8192*128*2 = 2 MiB
    float*    partial = (float*)(ws + (size_t)2 * 1024 * 1024);           // 8192*128*4 = 4 MiB
    float*    q       = (float*)(ws + (size_t)6 * 1024 * 1024);           // 8192*4     = 32 KiB
    float*    diag    = (float*)(ws + (size_t)6 * 1024 * 1024 + 65536);   // 4096*4     = 16 KiB

    saclr_normalize<<<1024, 256, 0, stream>>>(feats, Fn);                 // 8192 rows
    saclr_diag<<<512, 256, 0, stream>>>(feats, diag);                     // 4096 pairs
    saclr_gemm_exp<<<1032, 256, 0, stream>>>(Fn, partial);                // 8256 triangular wave-patches
    saclr_rowsum<<<1024, 256, 0, stream>>>(partial, q);                   // 8192 rows
    saclr_final<<<1, 256, 0, stream>>>(q, diag, s_inv, fidx, out);
}
